// MultiHeadAttention_8263517077733
// MI455X (gfx1250) — compile-verified, matrix operand order checked by reading, not run on hardware
//
#include <hip/hip_runtime.h>

typedef __bf16 bf16;
typedef __attribute__((ext_vector_type(16))) __bf16 v16bf;
typedef __attribute__((ext_vector_type(8)))  __bf16 v8bf;
typedef __attribute__((ext_vector_type(4)))  __bf16 v4bf;
typedef __attribute__((ext_vector_type(8)))  float  v8f;
typedef __attribute__((ext_vector_type(4)))  unsigned int u32x4;
typedef __attribute__((ext_vector_type(4)))  int i32x4;
typedef __attribute__((ext_vector_type(8)))  int i32x8;

#define DEVINL static __device__ __forceinline__
#define HAVE_TDM __has_builtin(__builtin_amdgcn_tensor_load_to_lds)

DEVINL v16bf cat8(v8bf lo, v8bf hi) {
  return __builtin_shufflevector(lo, hi, 0,1,2,3,4,5,6,7,8,9,10,11,12,13,14,15);
}

// -------------------------------------------------- Tensor Data Mover (TDM)
// 2-D bf16 tile load, global -> LDS.  dims/strides in elements, addrs in bytes.
// pad_cfg packs D# group1 bits [20] pad_enable, [24:22] pad_interval,
// [31:25] pad_amount.
DEVINL void tdm_load_2d(unsigned lds_addr, const void* gptr,
                        unsigned tile_d0, unsigned tile_d1,
                        unsigned tensor_d0, unsigned tensor_d1,
                        unsigned row_stride_elems, unsigned pad_cfg) {
#if HAVE_TDM
  unsigned long long ga = (unsigned long long)(size_t)gptr;
  u32x4 g0;
  g0[0] = 1u;                                              // count=1 (user D#)
  g0[1] = lds_addr;                                        // LDS byte address
  g0[2] = (unsigned)ga;                                    // global_addr[31:0]
  g0[3] = (unsigned)((ga >> 32) & 0x01FFFFFFu) | (2u << 30); // addr[56:32]|type=2
  i32x8 g1;
  g1[0] = (int)((1u << 16) | pad_cfg);                     // data_size=2B
  g1[1] = (int)((tensor_d0 & 0xFFFFu) << 16);              // tensor_dim0[15:0]
  g1[2] = (int)((tensor_d0 >> 16) | ((tensor_d1 & 0xFFFFu) << 16));
  g1[3] = (int)((tensor_d1 >> 16) | (tile_d0 << 16));      // tile_dim0
  g1[4] = (int)tile_d1;                                    // tile_dim1 (dim2=0)
  g1[5] = (int)row_stride_elems;                           // tensor_dim0_stride
  g1[6] = 0;
  g1[7] = 0;
  i32x4 z4 = {0, 0, 0, 0};
#if __clang_major__ >= 23
  i32x8 z8 = {0, 0, 0, 0, 0, 0, 0, 0};
  __builtin_amdgcn_tensor_load_to_lds(g0, g1, z4, z4, z8, 0);
#else
  __builtin_amdgcn_tensor_load_to_lds(g0, g1, z4, z4, 0);
#endif
#else
  (void)lds_addr; (void)gptr; (void)tile_d0; (void)tile_d1;
  (void)tensor_d0; (void)tensor_d1; (void)row_stride_elems; (void)pad_cfg;
#endif
}

DEVINL void tdm_wait0() {
#if HAVE_TDM
  __builtin_amdgcn_s_wait_tensorcnt(0);
#endif
}

DEVINL unsigned lds_byte_addr(const void* p) { return (unsigned)(size_t)p; }

// ---------------------------------------------------------------- fp32->bf16
__global__ __launch_bounds__(256)
void cvt_f32_bf16_x4(const float* __restrict__ s, bf16* __restrict__ d, int n4) {
  int i = blockIdx.x * 256 + threadIdx.x;
  if (i < n4) {
    float4 v = ((const float4*)s)[i];
    v4bf o;
    o[0] = (bf16)v.x; o[1] = (bf16)v.y; o[2] = (bf16)v.z; o[3] = (bf16)v.w;
    ((v4bf*)d)[i] = o;
  }
}

// -------------------------------------- fp32 [K][N] -> bf16 transposed [N][K]
__global__ __launch_bounds__(256)
void cvt_transpose_f32_bf16(const float* __restrict__ src, bf16* __restrict__ dst,
                            int K, int N) {
  __shared__ float tile[32][33];
  int k0 = blockIdx.y * 32, n0 = blockIdx.x * 32;
  int tx = threadIdx.x & 31, ty = threadIdx.x >> 5;   // 32 x 8
  #pragma unroll
  for (int i = 0; i < 4; ++i)
    tile[ty + i * 8][tx] = src[(size_t)(k0 + ty + i * 8) * N + n0 + tx];
  __syncthreads();
  #pragma unroll
  for (int i = 0; i < 4; ++i)
    dst[(size_t)(n0 + ty + i * 8) * K + k0 + tx] = (bf16)tile[tx][ty + i * 8];
}

// ---------------------------------------------------------------- WMMA GEMM
// D[M,N] = A[M,K] * Bt[N,K]^T;  A,Bt bf16 row-major; D bf16 or f32.
// 256 thr (8 waves, 4x2 grid), block tile 256x128, wave tile 64x64 (4x4 WMMA),
// BK=32, double-buffered TDM staging.
constexpr int BM = 256, BN = 128, BK = 32;
constexpr int AST = BK + 8;   // padded LDS row stride: 80B (TDM pad 16B / 64B)
// D# pad bits: enable | interval code 3 (16 DW = 64B) | amount code 3 (4 DW = 16B)
constexpr unsigned PAD_CFG = (1u << 20) | (3u << 22) | (3u << 25);

DEVINL void stage_rows_manual_1(bf16* dst, const bf16* src, int ld, int tid) {
  // 256 rows x 32 elems, 1 thread per row
  const v8bf* s = (const v8bf*)(src + (size_t)tid * ld);
  bf16* d = dst + tid * AST;
  *(v8bf*)(d)      = s[0];
  *(v8bf*)(d + 8)  = s[1];
  *(v8bf*)(d + 16) = s[2];
  *(v8bf*)(d + 24) = s[3];
}
DEVINL void stage_rows_manual_2(bf16* dst, const bf16* src, int ld, int tid) {
  // 128 rows x 32 elems, 2 threads per row
  int row = tid >> 1, seg = (tid & 1) * 16;
  const v8bf* s = (const v8bf*)(src + (size_t)row * ld + seg);
  *(v8bf*)(dst + row * AST + seg)     = s[0];
  *(v8bf*)(dst + row * AST + seg + 8) = s[1];
}

template <bool OUTF32>
__global__ __launch_bounds__(256)
void gemm_bf16_wmma(const bf16* __restrict__ A, const bf16* __restrict__ Bt,
                    void* __restrict__ D, int M, int N, int K) {
  __shared__ bf16 As[2][BM * AST];
  __shared__ bf16 Bs[2][BN * AST];
  const int tid  = threadIdx.x;
  const int lane = tid & 31, wave = tid >> 5;
  const int wr = wave >> 1, wc = wave & 1;            // 4x2 wave grid
  const int lhalf = lane >> 4, lmod = lane & 15;
  const int klo = lhalf * 8;
  const int m0 = blockIdx.y * BM, n0 = blockIdx.x * BN;

  v8f acc[4][4] = {};

#if HAVE_TDM
  if (wave == 0) {
    tdm_load_2d(lds_byte_addr(&As[0][0]), A  + (size_t)m0 * K, BK, BM, K, M, K, PAD_CFG);
    tdm_load_2d(lds_byte_addr(&Bs[0][0]), Bt + (size_t)n0 * K, BK, BN, K, N, K, PAD_CFG);
    tdm_wait0();
  }
#else
  stage_rows_manual_1(&As[0][0], A  + (size_t)m0 * K, K, tid);
  stage_rows_manual_2(&Bs[0][0], Bt + (size_t)n0 * K, K, tid);
#endif
  __syncthreads();

  int buf = 0;
  for (int k0 = 0; k0 < K; k0 += BK) {
    // prefetch next k-tile into the other buffer (overlaps with WMMA below)
    if (k0 + BK < K) {
#if HAVE_TDM
      if (wave == 0) {
        tdm_load_2d(lds_byte_addr(&As[buf ^ 1][0]), A  + (size_t)m0 * K + k0 + BK,
                    BK, BM, K, M, K, PAD_CFG);
        tdm_load_2d(lds_byte_addr(&Bs[buf ^ 1][0]), Bt + (size_t)n0 * K + k0 + BK,
                    BK, BN, K, N, K, PAD_CFG);
      }
#else
      stage_rows_manual_1(&As[buf ^ 1][0], A  + (size_t)m0 * K + k0 + BK, K, tid);
      stage_rows_manual_2(&Bs[buf ^ 1][0], Bt + (size_t)n0 * K + k0 + BK, K, tid);
#endif
    }

    const bf16* as = As[buf];
    const bf16* bs = Bs[buf];
    v16bf af[4], bfr[4];
    #pragma unroll
    for (int mt = 0; mt < 4; ++mt) {
      int m = wr * 64 + mt * 16 + lmod;
      af[mt] = cat8(*(const v8bf*)(as + m * AST + klo),
                    *(const v8bf*)(as + m * AST + klo + 16));
    }
    #pragma unroll
    for (int nt = 0; nt < 4; ++nt) {
      int n = wc * 64 + nt * 16 + lmod;
      bfr[nt] = cat8(*(const v8bf*)(bs + n * AST + klo),
                     *(const v8bf*)(bs + n * AST + klo + 16));
    }
    #pragma unroll
    for (int mt = 0; mt < 4; ++mt)
      #pragma unroll
      for (int nt = 0; nt < 4; ++nt)
        acc[mt][nt] = __builtin_amdgcn_wmma_f32_16x16x32_bf16(
            false, af[mt], false, bfr[nt], (short)0, acc[mt][nt], false, false);

#if HAVE_TDM
    if (wave == 0) tdm_wait0();
#endif
    __syncthreads();
    buf ^= 1;
  }

  #pragma unroll
  for (int mt = 0; mt < 4; ++mt)
    #pragma unroll
    for (int nt = 0; nt < 4; ++nt)
      #pragma unroll
      for (int r = 0; r < 8; ++r) {
        int row = m0 + wr * 64 + mt * 16 + r + 8 * lhalf;
        int col = n0 + wc * 64 + nt * 16 + lmod;
        float v = acc[mt][nt][r];
        if constexpr (OUTF32) ((float*)D)[(size_t)row * N + col] = v;
        else                  ((bf16*)D)[(size_t)row * N + col] = (bf16)v;
      }
}

// ------------------------------------------------------- flash attention
// qkv: [B,T,3C] bf16 (q|k|v), head h at dims [h*64, h*64+64).
// 128 thr (4 waves), 64 q rows per block, 16 per wave, KV step 32.
__global__ __launch_bounds__(128)
void flash_attn_wmma(const bf16* __restrict__ qkv, bf16* __restrict__ attn_out) {
  constexpr int C = 1024, C3 = 3072, T = 2048, HD = 64;
  __shared__ bf16 Ks[32 * 64];      // [key][d]  (TDM-staged)
  __shared__ bf16 Vt[64 * 32];      // [d][key]  (manual transpose)
  __shared__ bf16 Ps[4 * 16 * 32];  // per-wave P staging [row][key]

  const int tid = threadIdx.x, lane = tid & 31, wave = tid >> 5;
  const int lhalf = lane >> 4, lmod = lane & 15, klo = lhalf * 8;
  const int b = blockIdx.y >> 4, h = blockIdx.y & 15;
  const int q0 = blockIdx.x * 64;
  const int qb = q0 + wave * 16;

  const bf16* base = qkv + (size_t)b * T * C3;

  // Q fragments, A layout: lane row = lane%16, K-groups klo..+7 / klo+16..+23
  v16bf qa[2];
  {
    const bf16* qrow = base + (size_t)(qb + lmod) * C3 + h * HD;
    #pragma unroll
    for (int kc = 0; kc < 2; ++kc)
      qa[kc] = cat8(*(const v8bf*)(qrow + kc * 32 + klo),
                    *(const v8bf*)(qrow + kc * 32 + klo + 16));
  }

  v8f o[4] = {};
  float mi[8], li[8];
  #pragma unroll
  for (int r = 0; r < 8; ++r) { mi[r] = -1e30f; li[r] = 0.f; }

  const int jend = q0 + 64;
  for (int j0 = 0; j0 < jend; j0 += 32) {
    __syncthreads();
#if HAVE_TDM
    if (wave == 0)  // K tile 32 keys x 64 dims via TDM, row stride 3C
      tdm_load_2d(lds_byte_addr(Ks), base + (size_t)j0 * C3 + C + h * HD,
                  HD, 32, HD, T, C3, 0u);
#else
    #pragma unroll
    for (int it = 0; it < 2; ++it) {
      int idx = tid + it * 128;
      int kk = idx >> 3, seg = (idx & 7) * 8;
      *(v8bf*)(Ks + kk * 64 + seg) =
          *(const v8bf*)(base + (size_t)(j0 + kk) * C3 + C + h * HD + seg);
    }
#endif
    #pragma unroll
    for (int it = 0; it < 2; ++it) {      // stage V transposed
      int idx = tid + it * 128;
      int kk = idx >> 3, seg = (idx & 7) * 8;
      v8bf v = *(const v8bf*)(base + (size_t)(j0 + kk) * C3 + 2 * C + h * HD + seg);
      #pragma unroll
      for (int e = 0; e < 8; ++e) Vt[(seg + e) * 32 + kk] = v[e];
    }
#if HAVE_TDM
    if (wave == 0) tdm_wait0();
#endif
    __syncthreads();

    // S = Q K^T : 16 x 32 per wave (two 16x16 tiles, 2 k-steps over hd)
    v8f s[2];
    #pragma unroll
    for (int ct = 0; ct < 2; ++ct) {
      v8f a = {};
      #pragma unroll
      for (int kc = 0; kc < 2; ++kc) {
        int key = ct * 16 + lmod;
        int dlo = kc * 32 + klo;
        v16bf kb = cat8(*(const v8bf*)(Ks + key * 64 + dlo),
                        *(const v8bf*)(Ks + key * 64 + dlo + 16));
        a = __builtin_amdgcn_wmma_f32_16x16x32_bf16(false, qa[kc], false, kb,
                                                    (short)0, a, false, false);
      }
      s[ct] = a;
    }

    // scale + causal mask + online softmax (row lives in 16 lanes of a half)
    float mnew[8];
    #pragma unroll
    for (int r = 0; r < 8; ++r) {
      int qi = qb + r + 8 * lhalf;
      float v0 = s[0][r] * 0.125f;
      float v1 = s[1][r] * 0.125f;
      if (j0 + lmod      > qi) v0 = -1e30f;
      if (j0 + 16 + lmod > qi) v1 = -1e30f;
      s[0][r] = v0; s[1][r] = v1;
      float mx = fmaxf(v0, v1);
      #pragma unroll
      for (int off = 8; off; off >>= 1) mx = fmaxf(mx, __shfl_xor(mx, off, 32));
      mnew[r] = fmaxf(mi[r], mx);
    }
    #pragma unroll
    for (int r = 0; r < 8; ++r) {
      float alpha = __expf(mi[r] - mnew[r]);
      float p0 = __expf(s[0][r] - mnew[r]);
      float p1 = __expf(s[1][r] - mnew[r]);
      s[0][r] = p0; s[1][r] = p1;
      float rs = p0 + p1;
      #pragma unroll
      for (int off = 8; off; off >>= 1) rs += __shfl_xor(rs, off, 32);
      li[r] = li[r] * alpha + rs;
      mi[r] = mnew[r];
      #pragma unroll
      for (int n = 0; n < 4; ++n) o[n][r] *= alpha;
    }

    // C-layout -> A-layout for P via per-wave LDS (in-wave DS ops are ordered)
    bf16* ps = Ps + wave * 16 * 32;
    #pragma unroll
    for (int ct = 0; ct < 2; ++ct)
      #pragma unroll
      for (int r = 0; r < 8; ++r)
        ps[(r + 8 * lhalf) * 32 + ct * 16 + lmod] = (bf16)s[ct][r];
    v16bf pa = cat8(*(const v8bf*)(ps + lmod * 32 + klo),
                    *(const v8bf*)(ps + lmod * 32 + klo + 16));

    // O += P V
    #pragma unroll
    for (int n = 0; n < 4; ++n) {
      int d = n * 16 + lmod;
      v16bf vb = cat8(*(const v8bf*)(Vt + d * 32 + klo),
                      *(const v8bf*)(Vt + d * 32 + klo + 16));
      o[n] = __builtin_amdgcn_wmma_f32_16x16x32_bf16(false, pa, false, vb,
                                                     (short)0, o[n], false, false);
    }
  }

  #pragma unroll
  for (int r = 0; r < 8; ++r) {
    float inv = 1.f / li[r];
    int row = qb + r + 8 * lhalf;
    bf16* dst = attn_out + ((size_t)(b * T + row)) * C + h * HD;
    #pragma unroll
    for (int n = 0; n < 4; ++n) dst[n * 16 + lmod] = (bf16)(o[n][r] * inv);
  }
}

// ---------------------------------------------------------------- launcher
extern "C" void kernel_launch(void* const* d_in, const int* in_sizes, int n_in,
                              void* d_out, int out_size, void* d_ws, size_t ws_size,
                              hipStream_t stream) {
  (void)in_sizes; (void)n_in; (void)out_size; (void)ws_size;
  constexpr int Bb = 4, T = 2048, C = 1024, C3 = 3072, H = 16;
  constexpr size_t MT = (size_t)Bb * T;   // 8192 rows

  const float* x    = (const float*)d_in[0];
  const float* Wqkv = (const float*)d_in[1];
  const float* Wout = (const float*)d_in[2];
  float* out = (float*)d_out;

  char* ws = (char*)d_ws;
  bf16* xb    = (bf16*)ws;  ws += MT * C * 2;
  bf16* wqbT  = (bf16*)ws;  ws += (size_t)C * C3 * 2;   // [3C][C]
  bf16* wobT  = (bf16*)ws;  ws += (size_t)C * C * 2;    // [C][C]
  bf16* qkv   = (bf16*)ws;  ws += MT * C3 * 2;
  bf16* attn  = (bf16*)ws;

  // fp32 -> bf16 (x), fp32 -> bf16 transposed (weights, one-time)
  cvt_f32_bf16_x4<<<(int)(MT * C / 1024), 256, 0, stream>>>(x, xb, (int)(MT * C / 4));
  cvt_transpose_f32_bf16<<<dim3(C3 / 32, C / 32), 256, 0, stream>>>(Wqkv, wqbT, C, C3);
  cvt_transpose_f32_bf16<<<dim3(C / 32, C / 32), 256, 0, stream>>>(Wout, wobT, C, C);

  // qkv = x @ W_qkv   [8192,1024] x [1024,3072]
  gemm_bf16_wmma<false><<<dim3(C3 / BN, MT / BM), 256, 0, stream>>>(
      xb, wqbT, (void*)qkv, (int)MT, C3, C);

  // flash attention: grid (T/64, B*H)
  flash_attn_wmma<<<dim3(T / 64, Bb * H), 128, 0, stream>>>(qkv, attn);

  // out = attn @ W_out  [8192,1024] x [1024,1024] -> fp32
  gemm_bf16_wmma<true><<<dim3(C / BN, MT / BM), 256, 0, stream>>>(
      attn, wobT, (void*)out, (int)MT, C, C);
}
